// SpanElectraGeneratorPredictionHead_38800734552114
// MI455X (gfx1250) — matrix operand order repaired and checked by reading, not compiled, not run on hardware
//
#include <hip/hip_runtime.h>
#include <stdint.h>

// ---------------------------------------------------------------------------
// SpanElectra generator prediction head for MI455X (gfx1250, wave32, WMMA).
//   pack weights -> bf16 WMMA B-fragment images (L2 resident)
//   Ppair = gathered [lh|rh] @ W1[:1536]         (bf16 WMMA GEMM)
//   Ppos  = pos_emb @ W1[1536:]                  (tiny f32 kernel)
//   h1 = LN(gelu(Ppair[b]+Ppos[l]+b1))           (fused row kernel -> bf16)
//   h2 = LN(gelu(h1@W2 + b2))                    (WMMA GEMM + fused row kernel)
//   h3 = h2@Wp + bp                              (WMMA GEMM + bias->bf16)
//   out = h3 @ Wdec^T                            (WMMA GEMM, N=30528 padded)
// ---------------------------------------------------------------------------

typedef __attribute__((ext_vector_type(16))) __bf16 v16bf;
typedef __attribute__((ext_vector_type(8)))  float  v8f;

union __align__(16) FragBF {
    v16bf v;
    uint4 q[2];
    uint32_t u[8];
};

__device__ __forceinline__ uint32_t pack2bf(float a, float b) {
    uint32_t ua = __float_as_uint(a), ub = __float_as_uint(b);
    ua += 0x7FFFu + ((ua >> 16) & 1u);   // round-to-nearest-even to bf16
    ub += 0x7FFFu + ((ub >> 16) & 1u);
    return (ua >> 16) | (ub & 0xFFFF0000u);
}

// ---------------------------------------------------------------------------
// Pack a [K,N] f32 weight (optionally B[k][n] = src[n][k], i.e. transposed
// Wdec) into the per-lane bf16 B-fragment image for v_wmma_f32_16x16x32_bf16:
//   per (ntile, ktile): 32 lanes x 8 dwords; lane: half=lane/16, n=lane%16;
//   dword v holds K = {2v+16*half, +1} within the 32-row k-block.
// Linear dword index = ((nt*KT + kt)*32 + lane)*8 + v.
// ---------------------------------------------------------------------------
__global__ void pack_b_kernel(const float* __restrict__ W, uint32_t* __restrict__ out,
                              int K, int N, int ld, int nrealSrc, int transposed) {
    int idx = blockIdx.x * blockDim.x + threadIdx.x;
    int total = (K >> 1) * N;
    if (idx >= total) return;
    int KT   = K >> 5;
    int v    = idx & 7;
    int lane = (idx >> 3) & 31;
    int tile = idx >> 8;            // nt*KT + kt
    int kt   = tile % KT;
    int nt   = tile / KT;
    int half = lane >> 4;
    int n    = nt * 16 + (lane & 15);
    int k0   = kt * 32 + 2 * v + 16 * half;
    float f0, f1;
    if (!transposed) {
        f0 = W[(size_t)k0 * ld + n];
        f1 = W[(size_t)(k0 + 1) * ld + n];
    } else {
        bool ok = (n < nrealSrc);
        f0 = ok ? W[(size_t)n * ld + k0] : 0.f;
        f1 = ok ? W[(size_t)n * ld + k0 + 1] : 0.f;
    }
    out[idx] = pack2bf(f0, f1);
}

// Gather span endpoints into bf16 A rows: Apair[256][1536] = [lh | rh].
__global__ void gather_pairs_kernel(const float* __restrict__ hs, const int* __restrict__ pairs,
                                    uint32_t* __restrict__ Apair) {
    int idx = blockIdx.x * blockDim.x + threadIdx.x;   // 256*768 dwords
    if (idx >= 256 * 768) return;
    int r = idx / 768;
    int d = idx % 768;
    int c0 = d * 2;
    int side = (c0 >= 768) ? 1 : 0;
    int col = c0 - side * 768;
    int p = pairs[r * 2 + side];                        // pairs[bi][pi][side]
    const float* src = hs + ((size_t)(r >> 5) * 512 + p) * 768 + col;  // bi = r/32
    Apair[idx] = pack2bf(src[0], src[1]);
}

// Ppos[20][768] = pos_emb[20,200] @ W1[1536:1736, 768]   (tiny, plain f32)
__global__ void pos_proj_kernel(const float* __restrict__ pos_emb, const float* __restrict__ W1,
                                float* __restrict__ Ppos) {
    int idx = blockIdx.x * blockDim.x + threadIdx.x;
    if (idx >= 20 * 768) return;
    int l = idx / 768, n = idx % 768;
    float s = 0.f;
    for (int k = 0; k < 200; ++k)
        s += pos_emb[l * 200 + k] * W1[(size_t)(1536 + k) * 768 + n];
    Ppos[idx] = s;
}

// ---------------------------------------------------------------------------
// Tiled bf16 WMMA GEMM: C[M,LDC] f32 = A[M,K](bf16 rowmajor) x Bpacked[K,N].
// Block = 128 threads (4 waves). Macro tile 128(M) x 64(N); each wave owns a
// 32(M) x 64(N) tile: 2 A-frags x 4 B-frags -> 8 WMMAs per 32-wide k-step.
// B fragments stream from the L2-resident packed image (global_load_b128 x2).
//   K <= 128 : whole A block staged to LDS once -> barrier-free k-loop.
//   K >  128 : double-buffered LDS A tile, next k-step prefetched into regs
//              before current step's WMMAs -> one barrier per k-step.
// K / LDC / bounds-check are compile-time -> immediate-offset addressing.
// ---------------------------------------------------------------------------
template<int K, int LDC, bool CHECK_N>
__global__ __launch_bounds__(128)
void gemm_bf16_wmma(const uint32_t* __restrict__ A, const uint32_t* __restrict__ Bp,
                    float* __restrict__ C, int Nreal) {
    constexpr int KT  = K >> 5;   // 32-wide k-steps
    constexpr int Kd  = K >> 1;   // A row stride in dwords
    constexpr int QPR = K >> 3;   // uint4 per A row

    const int tid  = threadIdx.x;
    const int wave = tid >> 5;
    const int lane = tid & 31;
    const int half = lane >> 4;
    const int nl   = lane & 15;
    const int m0   = blockIdx.y * 128;
    const int n0   = blockIdx.x * 64;
    const int nt0  = n0 >> 4;
    const uint4* Aq = (const uint4*)A;

    v8f acc[2][4] = {};

    if constexpr (K <= 128) {
        // ---- stage entire 128 x K A block once; no barriers in k-loop ----
        constexpr int STRIDE = Kd + 4;                 // 68 dwords: 16B-aligned rows
        __shared__ __align__(16) uint32_t ldsA[128 * STRIDE];
        #pragma unroll
        for (int i = 0; i < QPR; ++i) {                // 128*QPR quads, QPR/thread
            int q = tid + i * 128;
            int r = q / QPR, c = q % QPR;
            *(uint4*)&ldsA[r * STRIDE + c * 4] = Aq[(size_t)(m0 + r) * QPR + c];
        }
        __syncthreads();
        #pragma unroll
        for (int kt = 0; kt < KT; ++kt) {
            FragBF a[2];
            #pragma unroll
            for (int mi = 0; mi < 2; ++mi) {
                int rowOff = (wave * 32 + mi * 16 + nl) * STRIDE + kt * 16;
                a[mi].q[0] = *(const uint4*)&ldsA[rowOff + 8 * half];
                a[mi].q[1] = *(const uint4*)&ldsA[rowOff + 8 * half + 4];
            }
            #pragma unroll
            for (int nt = 0; nt < 4; ++nt) {
                FragBF b;
                const uint4* pb = (const uint4*)&Bp[(((size_t)(nt0 + nt) * KT + kt) * 32 + lane) * 8];
                b.q[0] = pb[0];
                b.q[1] = pb[1];
                acc[0][nt] = __builtin_amdgcn_wmma_f32_16x16x32_bf16(
                    false, a[0].v, false, b.v, (short)0, acc[0][nt], false, false);
                acc[1][nt] = __builtin_amdgcn_wmma_f32_16x16x32_bf16(
                    false, a[1].v, false, b.v, (short)0, acc[1][nt], false, false);
            }
        }
    } else {
        // ---- double-buffered LDS A tile with register prefetch ----
        __shared__ __align__(16) uint32_t ldsA[2][128 * 20];
        uint4 pref[4];
        #pragma unroll
        for (int i = 0; i < 4; ++i) {                  // kt = 0 tile
            int q = tid + i * 128;
            int r = q >> 2, c = q & 3;
            pref[i] = Aq[(size_t)(m0 + r) * QPR + c];
        }
        #pragma unroll
        for (int i = 0; i < 4; ++i) {
            int q = tid + i * 128;
            int r = q >> 2, c = q & 3;
            *(uint4*)&ldsA[0][r * 20 + c * 4] = pref[i];
        }
        __syncthreads();
        int buf = 0;
        for (int kt = 0; kt < KT; ++kt) {
            if (kt + 1 < KT) {                         // prefetch next A tile
                #pragma unroll
                for (int i = 0; i < 4; ++i) {
                    int q = tid + i * 128;
                    int r = q >> 2, c = q & 3;
                    pref[i] = Aq[(size_t)(m0 + r) * QPR + (kt + 1) * 4 + c];
                }
            }
            FragBF a[2];
            #pragma unroll
            for (int mi = 0; mi < 2; ++mi) {
                int rowOff = (wave * 32 + mi * 16 + nl) * 20;
                a[mi].q[0] = *(const uint4*)&ldsA[buf][rowOff + 8 * half];
                a[mi].q[1] = *(const uint4*)&ldsA[buf][rowOff + 8 * half + 4];
            }
            #pragma unroll
            for (int nt = 0; nt < 4; ++nt) {
                FragBF b;
                const uint4* pb = (const uint4*)&Bp[(((size_t)(nt0 + nt) * KT + kt) * 32 + lane) * 8];
                b.q[0] = pb[0];
                b.q[1] = pb[1];
                acc[0][nt] = __builtin_amdgcn_wmma_f32_16x16x32_bf16(
                    false, a[0].v, false, b.v, (short)0, acc[0][nt], false, false);
                acc[1][nt] = __builtin_amdgcn_wmma_f32_16x16x32_bf16(
                    false, a[1].v, false, b.v, (short)0, acc[1][nt], false, false);
            }
            if (kt + 1 < KT) {                         // store prefetch to alt buffer
                #pragma unroll
                for (int i = 0; i < 4; ++i) {
                    int q = tid + i * 128;
                    int r = q >> 2, c = q & 3;
                    *(uint4*)&ldsA[buf ^ 1][r * 20 + c * 4] = pref[i];
                }
                __syncthreads();
                buf ^= 1;
            }
        }
    }

    // C/D layout: VGPR r -> M = r + 8*half, N = lane%16
    #pragma unroll
    for (int mi = 0; mi < 2; ++mi) {
        const size_t rbase = (size_t)(m0 + wave * 32 + mi * 16 + 8 * half);
        #pragma unroll
        for (int nt = 0; nt < 4; ++nt) {
            int col = n0 + nt * 16 + nl;
            if (!CHECK_N || col < Nreal) {
                #pragma unroll
                for (int r = 0; r < 8; ++r)
                    C[(rbase + r) * (size_t)LDC + col] = acc[mi][nt][r];
            }
        }
    }
}

// ---------------------------------------------------------------------------
// Fused bias + exact gelu + LayerNorm over a 768-wide row -> bf16 output.
// spanMode: x = Ppair[row/20] + Ppos[row%20] + bias ; else x = X[row] + bias.
// One block (128 threads, 6 elems each) per row.
// ---------------------------------------------------------------------------
__global__ __launch_bounds__(128)
void bias_gelu_ln_kernel(const float* __restrict__ X, const float* __restrict__ Ppos,
                         const float* __restrict__ bias, const float* __restrict__ gamma,
                         const float* __restrict__ beta, uint32_t* __restrict__ out,
                         int spanMode) {
    const int row = blockIdx.x;
    const int tid = threadIdx.x;
    const float* xr = spanMode ? (X + (size_t)(row / 20) * 768) : (X + (size_t)row * 768);
    const float* pr = spanMode ? (Ppos + (size_t)(row % 20) * 768) : nullptr;

    float g[6];
    float s = 0.f;
    const int nb = tid * 6;
    #pragma unroll
    for (int i = 0; i < 6; ++i) {
        int n = nb + i;
        float x = xr[n] + bias[n];
        if (pr) x += pr[n];
        float ge = 0.5f * x * (1.0f + erff(x * 0.70710678118654752440f));
        g[i] = ge;
        s += ge;
    }

    __shared__ float red[128];
    __shared__ float bcast;
    red[tid] = s; __syncthreads();
    for (int st = 64; st > 0; st >>= 1) {
        if (tid < st) red[tid] += red[tid + st];
        __syncthreads();
    }
    if (tid == 0) bcast = red[0] * (1.f / 768.f);
    __syncthreads();
    float mean = bcast;

    float vs = 0.f;
    #pragma unroll
    for (int i = 0; i < 6; ++i) { float d = g[i] - mean; vs += d * d; }
    red[tid] = vs; __syncthreads();
    for (int st = 64; st > 0; st >>= 1) {
        if (tid < st) red[tid] += red[tid + st];
        __syncthreads();
    }
    if (tid == 0) bcast = rsqrtf(red[0] * (1.f / 768.f) + 1e-12f);
    __syncthreads();
    float inv = bcast;

    uint32_t* orow = out + (size_t)row * 384;   // 768 bf16 = 384 dwords
    #pragma unroll
    for (int i = 0; i < 3; ++i) {
        int n0 = nb + 2 * i;
        float y0 = (g[2 * i]     - mean) * inv * gamma[n0]     + beta[n0];
        float y1 = (g[2 * i + 1] - mean) * inv * gamma[n0 + 1] + beta[n0 + 1];
        orow[tid * 3 + i] = pack2bf(y0, y1);
    }
}

// h3 = bf16(X + bp) elementwise (ncols = 128, even)
__global__ void bias_to_bf16_kernel(const float* __restrict__ X, const float* __restrict__ bias,
                                    uint32_t* __restrict__ out, int totalDwords, int ncols) {
    int idx = blockIdx.x * blockDim.x + threadIdx.x;
    if (idx >= totalDwords) return;
    int i0 = idx * 2;
    int n = i0 % ncols;
    out[idx] = pack2bf(X[i0] + bias[n], X[i0 + 1] + bias[n + 1]);
}

// ---------------------------------------------------------------------------
extern "C" void kernel_launch(void* const* d_in, const int* in_sizes, int n_in,
                              void* d_out, int out_size, void* d_ws, size_t ws_size,
                              hipStream_t stream) {
    (void)in_sizes; (void)n_in; (void)out_size; (void)ws_size;

    const float* hs    = (const float*)d_in[0];
    const int*   pairs = (const int*)  d_in[1];
    const float* pe    = (const float*)d_in[2];
    const float* W1    = (const float*)d_in[3];
    const float* b1    = (const float*)d_in[4];
    const float* g1    = (const float*)d_in[5];
    const float* be1   = (const float*)d_in[6];
    const float* W2    = (const float*)d_in[7];
    const float* b2    = (const float*)d_in[8];
    const float* g2    = (const float*)d_in[9];
    const float* be2   = (const float*)d_in[10];
    const float* Wp    = (const float*)d_in[11];
    const float* bp    = (const float*)d_in[12];
    const float* Wdec  = (const float*)d_in[13];
    float* out = (float*)d_out;

    const int M = 5120, VOCAB = 30522, NPAD = 30528;

    char* ws = (char*)d_ws;
    size_t off = 0;
    auto alloc = [&](size_t bytes) -> char* {
        char* p = ws + off;
        off = (off + bytes + 255) & ~(size_t)255;
        return p;
    };
    uint32_t* W1p   = (uint32_t*)alloc((size_t)1536 * 768 * 2);
    uint32_t* W2p   = (uint32_t*)alloc((size_t)768 * 768 * 2);
    uint32_t* Wpp   = (uint32_t*)alloc((size_t)768 * 128 * 2);
    uint32_t* Wdp   = (uint32_t*)alloc((size_t)128 * NPAD * 2);
    uint32_t* Apair = (uint32_t*)alloc((size_t)256 * 1536 * 2);
    float*    Ppair = (float*)   alloc((size_t)256 * 768 * 4);
    float*    Ppos  = (float*)   alloc((size_t)20 * 768 * 4);
    uint32_t* h1    = (uint32_t*)alloc((size_t)M * 768 * 2);
    float*    tmp   = (float*)   alloc((size_t)M * 768 * 4);   // h2pre, reused for h3pre
    uint32_t* h2    = (uint32_t*)alloc((size_t)M * 768 * 2);
    uint32_t* h3    = (uint32_t*)alloc((size_t)M * 128 * 2);

    auto cdiv = [](int a, int b) { return (a + b - 1) / b; };

    // weight packing (once per launch; cheap vs 625MB output)
    pack_b_kernel<<<cdiv(1536 * 768 / 2, 256), 256, 0, stream>>>(W1, W1p, 1536, 768, 768, 768, 0);
    pack_b_kernel<<<cdiv(768 * 768 / 2, 256), 256, 0, stream>>>(W2, W2p, 768, 768, 768, 768, 0);
    pack_b_kernel<<<cdiv(768 * 128 / 2, 256), 256, 0, stream>>>(Wp, Wpp, 768, 128, 128, 128, 0);
    pack_b_kernel<<<cdiv(128 * NPAD / 2, 256), 256, 0, stream>>>(Wdec, Wdp, 128, NPAD, 128, VOCAB, 1);

    pos_proj_kernel<<<cdiv(20 * 768, 256), 256, 0, stream>>>(pe, W1, Ppos);
    gather_pairs_kernel<<<cdiv(256 * 768, 256), 256, 0, stream>>>(hs, pairs, Apair);

    // GEMM1: Ppair[256,768] = Apair @ W1[:1536]
    gemm_bf16_wmma<1536, 768, false>
        <<<dim3(768 / 64, 256 / 128), 128, 0, stream>>>(Apair, W1p, Ppair, 768);
    // h1 = LN(gelu(Ppair[b] + Ppos[l] + b1))
    bias_gelu_ln_kernel<<<M, 128, 0, stream>>>(Ppair, Ppos, b1, g1, be1, h1, 1);

    // GEMM2 + epilogue: h2 = LN(gelu(h1 @ W2 + b2))
    gemm_bf16_wmma<768, 768, false>
        <<<dim3(768 / 64, M / 128), 128, 0, stream>>>(h1, W2p, tmp, 768);
    bias_gelu_ln_kernel<<<M, 128, 0, stream>>>(tmp, nullptr, b2, g2, be2, h2, 0);

    // GEMM3 + bias: h3 = bf16(h2 @ Wp + bp)
    gemm_bf16_wmma<768, 128, false>
        <<<dim3(128 / 64, M / 128), 128, 0, stream>>>(h2, Wpp, tmp, 128);
    bias_to_bf16_kernel<<<cdiv(M * 64, 256), 256, 0, stream>>>(tmp, bp, h3, M * 64, 128);

    // Decoder: out[5120, 30522] = h3 @ Wdec^T
    gemm_bf16_wmma<128, 30522, true>
        <<<dim3(NPAD / 64, M / 128), 128, 0, stream>>>(h3, Wdp, out, VOCAB);
}
